// ExpressionAttentionLayer_87093346828516
// MI455X (gfx1250) — matrix operand order repaired and checked by reading, not compiled, not run on hardware
//
#include <hip/hip_runtime.h>

// ExpressionAttentionLayer for MI455X (gfx1250, wave32, WMMA bf16)
// Pipeline:
//   1) proj_kernel      : q_fused/k_fused = Linear(cat(gene,expr)) -> bf16 [B*H,S,64]
//   2) vtrans_kernel    : V -> bf16 V^T [B*H,64,S]
//   3) wo_cvt_kernel    : Wo f32 -> bf16 [512,512]
//   4) flash_attn_kernel: online-softmax attention; K/V chunks staged to LDS with
//                         double-buffered GLOBAL_LOAD_ASYNC_TO_LDS_B128 (ASYNCcnt),
//                         split-barrier sync, v_wmma_f32_16x16x32_bf16 for QK^T / PV,
//                         P transposed via (bank-conflict-padded) LDS + s_wait_dscnt.
//   5) out_proj_kernel  : bf16 WMMA GEMM [8192x512] @ Wo^T + bo -> f32 out
//
// B=4, S=2048, H=8, HD=64, D=512.

#define B_SZ   4
#define S_LEN  2048
#define H_NUM  8
#define HD_SZ  64
#define D_SZ   512

// padded LDS row strides (elements) to break 64-bank conflicts
#define KROW   72    // 32 keys x 64 d   -> 144B rows
#define VROW   40    // 64 d   x 32 keys ->  80B rows
#define PROW   40    // 16 q   x 32 keys ->  80B rows

typedef __attribute__((ext_vector_type(16))) __bf16 v16bf;
typedef __attribute__((ext_vector_type(8)))  float  v8f;
typedef __attribute__((ext_vector_type(4)))  float  f32x4;
typedef __attribute__((ext_vector_type(4)))  int    i32x4;

#if __has_builtin(__builtin_amdgcn_global_load_async_to_lds_b128)
#define HAVE_ASYNC_LDS 1
#else
#define HAVE_ASYNC_LDS 0
#endif

// builtin wants: (int4 addrspace(1)*, int4 addrspace(3)*, imm offset, imm cpol)
#define AS1P(p) ((__attribute__((address_space(1))) i32x4*)(p))
#define AS3P(p) ((__attribute__((address_space(3))) i32x4*)(p))

#if __has_builtin(__builtin_amdgcn_s_wait_asynccnt)
#define WAIT_ASYNC(n) __builtin_amdgcn_s_wait_asynccnt(n)
#else
#define WAIT_ASYNC(n) asm volatile("s_wait_asynccnt %0" ::"i"(n) : "memory")
#endif

__device__ __forceinline__ void wg_barrier() {
#if __has_builtin(__builtin_amdgcn_s_barrier_signal) && \
    __has_builtin(__builtin_amdgcn_s_barrier_wait)
    __builtin_amdgcn_s_barrier_signal(-1);
    __builtin_amdgcn_s_barrier_wait(-1);
#else
    __syncthreads();
#endif
}

// ---------------------------------------------------------------------------
// 1) Fused projections: q_fused = [qg|qe] @ Wq^T + bq ; k_fused likewise.
// ---------------------------------------------------------------------------
__global__ __launch_bounds__(256) void proj_kernel(
    const float* __restrict__ Qg, const float* __restrict__ Kg,
    const float* __restrict__ Qe, const float* __restrict__ Ke,
    const float* __restrict__ Wk, const float* __restrict__ bk,
    const float* __restrict__ Wq, const float* __restrict__ bq,
    __bf16* __restrict__ qf, __bf16* __restrict__ kf)
{
    int idx = blockIdx.x * 256 + threadIdx.x;   // < 2^22
    int d  = idx & 63;
    int s  = (idx >> 6) & (S_LEN - 1);
    int bh = idx >> 17;                          // 0..31
    int b  = bh >> 3, h = bh & 7;

    size_t in_row = ((size_t)(b * S_LEN + s) * H_NUM + h) * HD_SZ;
    const float* wq = Wq + d * 128;
    const float* wk = Wk + d * 128;

    float aq = bq[d];
    float ak = bk[d];
#pragma unroll 8
    for (int j = 0; j < 64; ++j) {
        aq += wq[j] * Qg[in_row + j] + wq[64 + j] * Qe[in_row + j];
        ak += wk[j] * Kg[in_row + j] + wk[64 + j] * Ke[in_row + j];
    }
    size_t o = ((size_t)bh * S_LEN + s) * HD_SZ + d;
    qf[o] = (__bf16)aq;
    kf[o] = (__bf16)ak;
}

// ---------------------------------------------------------------------------
// 2) V transpose to [B*H, 64(d), S] bf16.
// ---------------------------------------------------------------------------
__global__ __launch_bounds__(256) void vtrans_kernel(
    const float* __restrict__ V, __bf16* __restrict__ vT)
{
    int idx = blockIdx.x * 256 + threadIdx.x;   // layout [B,S,H,HD]
    int d = idx & 63;
    int h = (idx >> 6) & 7;
    int s = (idx >> 9) & (S_LEN - 1);
    int b = idx >> 20;
    float v = V[idx];
    size_t o = ((size_t)(b * H_NUM + h) * HD_SZ + d) * S_LEN + s;
    vT[o] = (__bf16)v;
}

// ---------------------------------------------------------------------------
// 3) Wo f32 -> bf16.
// ---------------------------------------------------------------------------
__global__ __launch_bounds__(256) void wo_cvt_kernel(
    const float* __restrict__ Wo, __bf16* __restrict__ Wob)
{
    int idx = blockIdx.x * 256 + threadIdx.x;   // 262144
    Wob[idx] = (__bf16)Wo[idx];
}

// ---------------------------------------------------------------------------
// cooperative K/V chunk staging: 256 threads copy 16B each of K (32x64 bf16)
// and V^T (64x32 bf16) into padded LDS tiles, via async DMA when available.
// ---------------------------------------------------------------------------
__device__ __forceinline__ void stage_kv(const __bf16* kbase, const __bf16* vbase,
                                         int k0, __bf16* sK, __bf16* sV, int tid)
{
    const __bf16* gK = kbase + (size_t)(k0 + (tid >> 3)) * HD_SZ + (tid & 7) * 8;
    __bf16*       dK = sK + (tid >> 3) * KROW + (tid & 7) * 8;
    const __bf16* gV = vbase + (size_t)(tid >> 2) * S_LEN + k0 + (tid & 3) * 8;
    __bf16*       dV = sV + (tid >> 2) * VROW + (tid & 3) * 8;
#if HAVE_ASYNC_LDS
    __builtin_amdgcn_global_load_async_to_lds_b128(AS1P(gK), AS3P(dK), 0, 0);
    __builtin_amdgcn_global_load_async_to_lds_b128(AS1P(gV), AS3P(dV), 0, 0);
#else
    *(f32x4*)dK = *(const f32x4*)gK;
    *(f32x4*)dV = *(const f32x4*)gV;
#endif
}

// ---------------------------------------------------------------------------
// 4) Flash attention. One wave = one 16-query tile; 8 waves/block share one
//    (b,h): K/V chunks staged once per block (8x traffic cut), double-buffered.
//    grid = B*H * (S/128) = 512 blocks of 256 threads.
// ---------------------------------------------------------------------------
__global__ __launch_bounds__(256) void flash_attn_kernel(
    const __bf16* __restrict__ qf, const __bf16* __restrict__ kf,
    const __bf16* __restrict__ vT, const float* __restrict__ Mmask,
    __bf16* __restrict__ Obf)
{
    __shared__ __bf16 sK[2][32 * KROW];          // 2 x 4.5 KB
    __shared__ __bf16 sV[2][64 * VROW];          // 2 x 5.0 KB
    __shared__ __bf16 pbuf[8][16 * PROW];        // 10 KB

    const int tid    = threadIdx.x;
    const int lane   = tid & 31;
    const int wave   = tid >> 5;
    const int lane16 = lane & 15;
    const int hi16   = lane >> 4;                // 0: lanes 0-15, 1: lanes 16-31

    const int bh   = blockIdx.x >> 4;            // 0..31
    const int qblk = blockIdx.x & 15;
    const int b    = bh >> 3, h = bh & 7;
    const int q0   = qblk * 128 + wave * 16;

    const __bf16* qbase = qf + (size_t)bh * S_LEN * HD_SZ;
    const __bf16* kbase = kf + (size_t)bh * S_LEN * HD_SZ;
    const __bf16* vbase = vT + (size_t)bh * HD_SZ * S_LEN;
    const float*  mbase = Mmask + (size_t)b * S_LEN * S_LEN;

    // ---- Q fragments (A layout, 16x32 per d-chunk); row = lane16.
    v16bf aq0, aq1;
    {
        const __bf16* qrow = qbase + (size_t)(q0 + lane16) * HD_SZ;
#pragma unroll
        for (int i = 0; i < 16; ++i) {
            int v = i >> 1, p = i & 1;
            int d = ((v >> 2) << 4) + (hi16 << 3) + ((v & 3) << 1) + p;
            aq0[i] = qrow[d];
            aq1[i] = qrow[32 + d];
        }
    }

    v8f acc[4] = {};                             // O tile: 16 q x 64 d
    float mrow[8], lrow[8];
#pragma unroll
    for (int r = 0; r < 8; ++r) { mrow[r] = -3.0e38f; lrow[r] = 0.0f; }

    __bf16* myp = &pbuf[wave][0];
    const float scale = 0.125f;                  // 1/sqrt(64)

    stage_kv(kbase, vbase, 0, sK[0], sV[0], tid);  // prime buffer 0

    for (int k0 = 0; k0 < S_LEN; k0 += 32) {
        const int  cur  = (k0 >> 5) & 1;
        const bool more = (k0 + 32) < S_LEN;
        if (more) stage_kv(kbase, vbase, k0 + 32, sK[cur ^ 1], sV[cur ^ 1], tid);
#if HAVE_ASYNC_LDS
        if (more) { WAIT_ASYNC(2); } else { WAIT_ASYNC(0); }  // in-order: cur done
#else
        asm volatile("s_wait_dscnt 0" ::: "memory");          // sync-copy fallback
#endif
        wg_barrier();                            // everyone's cur chunk resident

        const __bf16* sKc = &sK[cur][0];
        const __bf16* sVc = &sV[cur][0];

        // -------- logits: two 16x16 tiles, K(d)=64 -> 2 wmma each
        v8f st[2];
#pragma unroll
        for (int u = 0; u < 2; ++u) {
            v8f s_ = {};
            const __bf16* krow = sKc + (u * 16 + lane16) * KROW + (hi16 << 3);
            v16bf bk0, bk1;
#pragma unroll
            for (int i = 0; i < 16; ++i) { bk0[i] = krow[(i & 7) + ((i >> 3) << 4)]; bk1[i] = krow[32 + (i & 7) + ((i >> 3) << 4)]; }
            s_ = __builtin_amdgcn_wmma_f32_16x16x32_bf16(false, aq0, false, bk0,
                                                         (short)0, s_, false, false);
            s_ = __builtin_amdgcn_wmma_f32_16x16x32_bf16(false, aq1, false, bk1,
                                                         (short)0, s_, false, false);
            st[u] = s_;
        }

        // prefetch next mask chunk (global_prefetch_b8)
        if (more) {
            __builtin_prefetch(
                &mbase[(size_t)(q0 + (hi16 << 3)) * S_LEN + k0 + 32 + lane16], 0, 1);
        }

        // -------- row max over the 32 new keys (reduce across 16-lane half)
        float rmax[8];
#pragma unroll
        for (int r = 0; r < 8; ++r)
            rmax[r] = fmaxf(st[0][r], st[1][r]) * scale;
#pragma unroll
        for (int off = 1; off < 16; off <<= 1) {
#pragma unroll
            for (int r = 0; r < 8; ++r)
                rmax[r] = fmaxf(rmax[r], __shfl_xor(rmax[r], off, 32));
        }

        // -------- online rescale
        float corr[8];
#pragma unroll
        for (int r = 0; r < 8; ++r) {
            float mn = fmaxf(mrow[r], rmax[r]);
            corr[r]  = __expf(mrow[r] - mn);
            mrow[r]  = mn;
            lrow[r] *= corr[r];
        }
#pragma unroll
        for (int t = 0; t < 4; ++t) {
#pragma unroll
            for (int r = 0; r < 8; ++r) acc[t][r] *= corr[r];
        }

        // -------- probs; denominator uses UNMASKED p, numerator uses p * M
        float rsum[8];
#pragma unroll
        for (int r = 0; r < 8; ++r) rsum[r] = 0.0f;
#pragma unroll
        for (int u = 0; u < 2; ++u) {
            int key = k0 + u * 16 + lane16;
#pragma unroll
            for (int r = 0; r < 8; ++r) {
                int q    = q0 + r + (hi16 << 3);
                float p  = __expf(st[u][r] * scale - mrow[r]);
                rsum[r] += p;
                float mv = mbase[(size_t)q * S_LEN + key];
                myp[(r + (hi16 << 3)) * PROW + u * 16 + lane16] = (__bf16)(p * mv);
            }
        }
#pragma unroll
        for (int off = 1; off < 16; off <<= 1) {
#pragma unroll
            for (int r = 0; r < 8; ++r)
                rsum[r] += __shfl_xor(rsum[r], off, 32);
        }
#pragma unroll
        for (int r = 0; r < 8; ++r) lrow[r] += rsum[r];

        // make private P stores visible before re-reading in A-layout
        asm volatile("s_wait_dscnt 0" ::: "memory");

        // -------- P fragment (A layout 16x32) from LDS
        v16bf pf;
        {
            const __bf16* prow = myp + lane16 * PROW;
#pragma unroll
            for (int i = 0; i < 16; ++i) {
                int v = i >> 1, p = i & 1;
                int kk = ((v >> 2) << 4) + (hi16 << 3) + ((v & 3) << 1) + p;
                pf[i] = prow[kk];
            }
        }

        // -------- PV: 4 output tiles of 16x16
#pragma unroll
        for (int t = 0; t < 4; ++t) {
            const __bf16* vrow = sVc + (t * 16 + lane16) * VROW + (hi16 << 3);
            v16bf bv;
#pragma unroll
            for (int i = 0; i < 16; ++i) bv[i] = vrow[(i & 7) + ((i >> 3) << 4)];
            acc[t] = __builtin_amdgcn_wmma_f32_16x16x32_bf16(false, pf, false, bv,
                                                             (short)0, acc[t],
                                                             false, false);
        }

        wg_barrier();   // all waves done reading cur before it is overwritten
    }

    // -------- epilogue: divide by full softmax denominator, emit bf16 [B,S,D]
    float inv[8];
#pragma unroll
    for (int r = 0; r < 8; ++r) {
#if __has_builtin(__builtin_amdgcn_rcpf)
        inv[r] = __builtin_amdgcn_rcpf(lrow[r]);
#else
        inv[r] = 1.0f / lrow[r];
#endif
    }
#pragma unroll
    for (int t = 0; t < 4; ++t) {
#pragma unroll
        for (int r = 0; r < 8; ++r) {
            int q = q0 + r + (hi16 << 3);
            size_t o = (size_t)(b * S_LEN + q) * D_SZ + h * HD_SZ + t * 16 + lane16;
            Obf[o] = (__bf16)(acc[t][r] * inv[r]);
        }
    }
}

// ---------------------------------------------------------------------------
// 5) Output projection: out[m,n] = sum_k O[m,k]*Wo[n,k] + bo[n]
//    M=8192, N=512, K=512. One wave -> 16x64 tile; 8 waves/block; 512 blocks.
// ---------------------------------------------------------------------------
__global__ __launch_bounds__(256) void out_proj_kernel(
    const __bf16* __restrict__ A, const __bf16* __restrict__ Wob,
    const float* __restrict__ bo, float* __restrict__ out)
{
    const int lane   = threadIdx.x & 31;
    const int wave   = threadIdx.x >> 5;
    const int lane16 = lane & 15;
    const int hi16   = lane >> 4;

    int tile = blockIdx.x * 8 + wave;            // 0..4095
    int m0   = (tile >> 3) * 16;                 // 512 row-tiles
    int n0   = (tile & 7) * 64;                  // 8 col-groups

    v8f acc[4] = {};
    for (int kc = 0; kc < D_SZ; kc += 32) {
        v16bf af;
        const __bf16* arow = A + (size_t)(m0 + lane16) * D_SZ + kc;
#pragma unroll
        for (int i = 0; i < 16; ++i) {
            int v = i >> 1, p = i & 1;
            int kk = ((v >> 2) << 4) + (hi16 << 3) + ((v & 3) << 1) + p;
            af[i] = arow[kk];
        }
#pragma unroll
        for (int t = 0; t < 4; ++t) {
            const __bf16* wrow =
                Wob + (size_t)(n0 + t * 16 + lane16) * D_SZ + kc + (hi16 << 3);
            v16bf bf_;
#pragma unroll
            for (int i = 0; i < 16; ++i) bf_[i] = wrow[(i & 7) + ((i >> 3) << 4)];
            acc[t] = __builtin_amdgcn_wmma_f32_16x16x32_bf16(false, af, false, bf_,
                                                             (short)0, acc[t],
                                                             false, false);
        }
    }
#pragma unroll
    for (int t = 0; t < 4; ++t) {
#pragma unroll
        for (int r = 0; r < 8; ++r) {
            int m = m0 + r + (hi16 << 3);
            int n = n0 + t * 16 + lane16;
            out[(size_t)m * D_SZ + n] = acc[t][r] + bo[n];
        }
    }
}

// ---------------------------------------------------------------------------
extern "C" void kernel_launch(void* const* d_in, const int* in_sizes, int n_in,
                              void* d_out, int out_size, void* d_ws, size_t ws_size,
                              hipStream_t stream)
{
    const float* Qg = (const float*)d_in[0];
    const float* Kg = (const float*)d_in[1];
    const float* Qe = (const float*)d_in[2];
    const float* Ke = (const float*)d_in[3];
    const float* Ve = (const float*)d_in[4];
    const float* Mm = (const float*)d_in[5];
    const float* Wk = (const float*)d_in[6];
    const float* bk = (const float*)d_in[7];
    const float* Wq = (const float*)d_in[8];
    const float* bq = (const float*)d_in[9];
    const float* Wo = (const float*)d_in[10];
    const float* bo = (const float*)d_in[11];
    float* out = (float*)d_out;

    // workspace carve-up (~33 MB): qf, kf, vT, O, Wo_bf16
    char* ws = (char*)d_ws;
    const size_t QKV = (size_t)B_SZ * H_NUM * S_LEN * HD_SZ;     // 4.19M elems
    __bf16* qf  = (__bf16*)ws;  ws += QKV * 2;
    __bf16* kf  = (__bf16*)ws;  ws += QKV * 2;
    __bf16* vt  = (__bf16*)ws;  ws += QKV * 2;
    __bf16* Ob  = (__bf16*)ws;  ws += (size_t)B_SZ * S_LEN * D_SZ * 2;
    __bf16* Wob = (__bf16*)ws;  ws += (size_t)D_SZ * D_SZ * 2;

    proj_kernel      <<<16384, 256, 0, stream>>>(Qg, Kg, Qe, Ke, Wk, bk, Wq, bq, qf, kf);
    vtrans_kernel    <<<16384, 256, 0, stream>>>(Ve, vt);
    wo_cvt_kernel    <<<1024,  256, 0, stream>>>(Wo, Wob);
    flash_attn_kernel<<<512,   256, 0, stream>>>(qf, kf, vt, Mm, Ob);
    out_proj_kernel  <<<512,   256, 0, stream>>>(Ob, Wob, bo, out);
}
